// MLP_30485677867761
// MI455X (gfx1250) — matrix operand order repaired and checked by reading, not compiled, not run on hardware
//
#include <hip/hip_runtime.h>

// ---------------------------------------------------------------------------
// MLP forward for MI455X (gfx1250, wave32, WMMA).
//   h = BN_ReLU(x @ W0^T + b0); 4x { h = BN_ReLU(h @ Wh^T + bh) }; out = h @ Wout^T + bout
// bf16 activations/weights, fp32 WMMA accumulate, BN fused into next layer's
// A-tile staging. Weight tiles use gfx1250 async global->LDS DMA when the
// toolchain exposes the builtins. Deterministic (no float atomics).
// ---------------------------------------------------------------------------

typedef __bf16 bf16_t;
typedef bf16_t v8bf  __attribute__((ext_vector_type(8)));
typedef bf16_t v16bf __attribute__((ext_vector_type(16)));
typedef float  v8f   __attribute__((ext_vector_type(8)));
typedef int    v4i   __attribute__((vector_size(16)));   // matches builtin param type

#if defined(__has_builtin)
#if __has_builtin(__builtin_amdgcn_global_load_async_to_lds_b128) && \
    __has_builtin(__builtin_amdgcn_s_wait_asynccnt)
#define USE_ASYNC_LDS 1
#endif
#endif
#ifndef USE_ASYNC_LDS
#define USE_ASYNC_LDS 0
#endif

#define EMB     2048
#define NROW    16384
#define NLAYER  5
#define EPS_BN  1e-5f

// GEMM tiling
#define BM   128
#define BN   128
#define BK   32
#define LDSP 40          // padded LDS row (bf16 elems): 80B stride -> conflict free, 16B aligned

// BN stats reduction
#define RCHUNK  256
#define NCHUNKS (NROW / RCHUNK)   // 64

// ---------------------------------------------------------------------------
// fp32 -> bf16 convert
// ---------------------------------------------------------------------------
__global__ __launch_bounds__(256)
void mlp_cvt_bf16(const float* __restrict__ src, bf16_t* __restrict__ dst, int n) {
    int i = blockIdx.x * 256 + threadIdx.x;
    if (i < n) dst[i] = (bf16_t)src[i];
}

// ---------------------------------------------------------------------------
// C[N,EMB](bf16, pre-BN) = act(A)[N,K] @ W[EMB,K]^T + bias
// If aScale != null, A elements get fused BN+ReLU of the *previous* layer:
//   a' = max(0, a*aScale[k] + aShift[k])  (k == previous layer's column index)
// W tiles are DMA'd global->LDS asynchronously (ASYNCcnt); A tiles go through
// VGPRs because of the fused BN transform.
// ---------------------------------------------------------------------------
__global__ __launch_bounds__(256)
void mlp_gemm_bf16(const bf16_t* __restrict__ A, const bf16_t* __restrict__ W,
                   const float* __restrict__ bias,
                   const float* __restrict__ aScale, const float* __restrict__ aShift,
                   bf16_t* __restrict__ C, int K)
{
    __shared__ __align__(16) bf16_t As[2][BM * LDSP];
    __shared__ __align__(16) bf16_t Bs[2][BN * LDSP];

    const int tid  = threadIdx.x;
    const int lane = tid & 31;
    const int wave = tid >> 5;          // 0..7
    const int wRow = (wave & 3) * 32;   // 4 waves along M
    const int wCol = (wave >> 2) * 64;  // 2 waves along N
    const int rowBase = blockIdx.x * BM;
    const int colBase = blockIdx.y * BN;

    v8f acc[2][4] = {};

    const bf16_t* aBase = A + (size_t)rowBase * K;
    const bf16_t* wBase = W + (size_t)colBase * K;

    auto loadTile = [&](int buf, int kt) {
        const int k0 = kt * BK;
#pragma unroll
        for (int it = 0; it < 2; ++it) {
            const int idx = tid + it * 256;   // 0..511 -> 512 x 16B = 8KB per matrix
            const int r   = idx >> 2;         // row 0..127
            const int c8  = (idx & 3) * 8;    // bf16 column chunk
            // B (weights): straight copy -> async DMA global->LDS, no VGPR staging
#if USE_ASYNC_LDS
            __builtin_amdgcn_global_load_async_to_lds_b128(
                (v4i*)(wBase + (size_t)r * K + k0 + c8),      // global src (VADDR)
                (v4i*)&Bs[buf][r * LDSP + c8],                // LDS dst  (VDST)
                0, 0);
#else
            uint4 vb = *(const uint4*)(wBase + (size_t)r * K + k0 + c8);
            *(uint4*)&Bs[buf][r * LDSP + c8] = vb;
#endif
            // A (activations): VGPR path with fused BN + ReLU of previous layer
            uint4 va = *(const uint4*)(aBase + (size_t)r * K + k0 + c8);
            if (aScale) {
                alignas(16) bf16_t t[8];
                *(uint4*)t = va;
                const float* s = aScale + k0 + c8;
                const float* h = aShift + k0 + c8;
#pragma unroll
                for (int j = 0; j < 8; ++j) {
                    float v = (float)t[j] * s[j] + h[j];
                    t[j] = (bf16_t)fmaxf(v, 0.f);
                }
                va = *(const uint4*)t;
            }
            *(uint4*)&As[buf][r * LDSP + c8] = va;
        }
    };

    auto tileBarrier = [&]() {
#if USE_ASYNC_LDS
        __builtin_amdgcn_s_wait_asynccnt(0);  // ASYNCcnt not covered by barrier
#endif
        __syncthreads();                      // covers DScnt + workgroup barrier
    };

    loadTile(0, 0);
    tileBarrier();

    // fragment addressing used here (wave32, contiguous k = 16*half + s, not ISA table order):
    //  A 16x32 bf16: lane L<16 holds row M=L, K=0..15; lane L+16: K=16..31
    //  B 32x16 bf16: lane L<16 holds col N=L, K=0..15; lane L+16: K=16..31
    const int lrow = lane & 15;
    const int hi   = lane >> 4;
    const int aOff = hi * 16;
    const int bOff = hi * 16;
    const int nk   = K / BK;

    int cur = 0;
    for (int kt = 0; kt < nk; ++kt) {
        if (kt + 1 < nk) loadTile(cur ^ 1, kt + 1);

        v16bf bfr[4];
#pragma unroll
        for (int jn = 0; jn < 4; ++jn) {
            const bf16_t* bp = &Bs[cur][(wCol + jn * 16 + lrow) * LDSP + bOff];
            v8bf blo = *(const v8bf*)bp;
            v8bf bhi = *(const v8bf*)(bp + 8);
            bfr[jn] = __builtin_shufflevector(blo, bhi,
                       0,1,2,3,4,5,6,7,8,9,10,11,12,13,14,15);
        }
#pragma unroll
        for (int im = 0; im < 2; ++im) {
            const bf16_t* ap = &As[cur][(wRow + im * 16 + lrow) * LDSP + aOff];
            v8bf alo = *(const v8bf*)ap;
            v8bf ahi = *(const v8bf*)(ap + 8);
            v16bf afr = __builtin_shufflevector(alo, ahi,
                         0,1,2,3,4,5,6,7,8,9,10,11,12,13,14,15);
#pragma unroll
            for (int jn = 0; jn < 4; ++jn) {
                acc[im][jn] = __builtin_amdgcn_wmma_f32_16x16x32_bf16(
                    false, afr, false, bfr[jn], (short)0, acc[im][jn], false, false);
            }
        }
        tileBarrier();
        cur ^= 1;
    }

    // Epilogue: +bias, store pre-BN bf16.
    // C/D layout: VGPR r -> M = r + (lane>=16 ? 8 : 0); N = lane&15
#pragma unroll
    for (int jn = 0; jn < 4; ++jn) {
        const int col = colBase + wCol + jn * 16 + lrow;
        const float bv = bias[col];
#pragma unroll
        for (int im = 0; im < 2; ++im) {
            const int rb = rowBase + wRow + im * 16 + hi * 8;
#pragma unroll
            for (int r = 0; r < 8; ++r) {
                C[(size_t)(rb + r) * EMB + col] = (bf16_t)(acc[im][jn][r] + bv);
            }
        }
    }
}

// ---------------------------------------------------------------------------
// BN stats, stage 1: per-(row-chunk, column) partial sum / sumsq (deterministic)
// grid = (EMB/128, NCHUNKS), block = 128.  H was just written -> L2 resident.
// ---------------------------------------------------------------------------
__global__ __launch_bounds__(128)
void mlp_bn_partial(const bf16_t* __restrict__ H,
                    float* __restrict__ psum, float* __restrict__ psq)
{
    const int col = blockIdx.x * 128 + threadIdx.x;
    const int r0  = blockIdx.y * RCHUNK;
    float s = 0.f, q = 0.f;
    for (int r = 0; r < RCHUNK; ++r) {
        float v = (float)H[(size_t)(r0 + r) * EMB + col];
        s += v;
        q += v * v;
    }
    psum[blockIdx.y * EMB + col] = s;
    psq [blockIdx.y * EMB + col] = q;
}

// ---------------------------------------------------------------------------
// BN stats, stage 2: fold partials (fixed order), emit per-column scale/shift
//   a = gamma * rsqrt(var+eps) ; c = beta - mu*a
// ---------------------------------------------------------------------------
__global__ __launch_bounds__(256)
void mlp_bn_finalize(const float* __restrict__ psum, const float* __restrict__ psq,
                     const float* __restrict__ gamma, const float* __restrict__ beta,
                     float* __restrict__ a, float* __restrict__ c)
{
    const int col = blockIdx.x * 256 + threadIdx.x;
    float s = 0.f, q = 0.f;
    for (int i = 0; i < NCHUNKS; ++i) {
        s += psum[i * EMB + col];
        q += psq [i * EMB + col];
    }
    const float inv_n = 1.0f / (float)NROW;
    const float mu  = s * inv_n;
    const float var = q * inv_n - mu * mu;
    const float sc  = gamma[col] * rsqrtf(var + EPS_BN);
    a[col] = sc;
    c[col] = beta[col] - mu * sc;
}

// ---------------------------------------------------------------------------
// out[n] = sum_e BN_ReLU(H[n,e]) * Wout[e] + bout   (wave per row)
// ---------------------------------------------------------------------------
__global__ __launch_bounds__(256)
void mlp_out_gemv(const bf16_t* __restrict__ H,
                  const float* __restrict__ aScale, const float* __restrict__ aShift,
                  const float* __restrict__ Wout, const float* __restrict__ bout,
                  float* __restrict__ out)
{
    const int row  = blockIdx.x * 8 + (threadIdx.x >> 5);
    const int lane = threadIdx.x & 31;
    const bf16_t* hp = H + (size_t)row * EMB;
    float s = 0.f;
    for (int e = lane; e < EMB; e += 32) {
        float v = (float)hp[e] * aScale[e] + aShift[e];
        s += fmaxf(v, 0.f) * Wout[e];
    }
#pragma unroll
    for (int off = 16; off > 0; off >>= 1) s += __shfl_down(s, off);
    if (lane == 0) out[row] = s + bout[0];
}

// ---------------------------------------------------------------------------
// Host launch
// ---------------------------------------------------------------------------
extern "C" void kernel_launch(void* const* d_in, const int* in_sizes, int n_in,
                              void* d_out, int out_size, void* d_ws, size_t ws_size,
                              hipStream_t stream) {
    (void)in_sizes; (void)n_in; (void)out_size; (void)ws_size;
    const float* x     = (const float*)d_in[0];   // [16384,128]
    const float* W0    = (const float*)d_in[1];   // [2048,128]
    const float* b0    = (const float*)d_in[2];   // [2048]
    const float* Wh    = (const float*)d_in[3];   // [4,2048,2048]
    const float* bh    = (const float*)d_in[4];   // [4,2048]
    const float* gamma = (const float*)d_in[5];   // [5,2048]
    const float* beta  = (const float*)d_in[6];   // [5,2048]
    const float* Wout  = (const float*)d_in[7];   // [1,2048]
    const float* bout  = (const float*)d_in[8];   // [1]
    float* out = (float*)d_out;                   // [16384,1]

    // workspace carve-up (256B aligned slabs)
    char* ws = (char*)d_ws;
    size_t off = 0;
    auto carve = [&](size_t bytes) -> void* {
        void* p = ws + off;
        off = (off + bytes + 255) & ~(size_t)255;
        return p;
    };
    const size_t W0_ELEMS = (size_t)EMB * 128;
    const size_t WH_ELEMS = (size_t)4 * EMB * EMB;
    bf16_t* xbf  = (bf16_t*)carve((size_t)NROW * 128 * sizeof(bf16_t));
    bf16_t* Wbf  = (bf16_t*)carve((W0_ELEMS + WH_ELEMS) * sizeof(bf16_t));
    bf16_t* H0   = (bf16_t*)carve((size_t)NROW * EMB * sizeof(bf16_t));
    bf16_t* H1   = (bf16_t*)carve((size_t)NROW * EMB * sizeof(bf16_t));
    float*  psum = (float*)carve((size_t)NCHUNKS * EMB * sizeof(float));
    float*  psq  = (float*)carve((size_t)NCHUNKS * EMB * sizeof(float));
    float*  sa   = (float*)carve((size_t)NLAYER * EMB * sizeof(float));
    float*  sc   = (float*)carve((size_t)NLAYER * EMB * sizeof(float));

    // fp32 -> bf16 conversions (every call; deterministic)
    {
        int n = NROW * 128;
        mlp_cvt_bf16<<<(n + 255) / 256, 256, 0, stream>>>(x, xbf, n);
        n = (int)W0_ELEMS;
        mlp_cvt_bf16<<<(n + 255) / 256, 256, 0, stream>>>(W0, Wbf, n);
        n = (int)WH_ELEMS;
        mlp_cvt_bf16<<<(n + 255) / 256, 256, 0, stream>>>(Wh, Wbf + W0_ELEMS, n);
    }

    const dim3 gemmGrid(NROW / BM, EMB / BN);   // (128, 16)
    const dim3 statGrid(EMB / 128, NCHUNKS);    // (16, 64)

    for (int i = 0; i < NLAYER; ++i) {
        const bf16_t* Ain   = (i == 0) ? xbf : ((i & 1) ? H0 : H1); // ping-pong
        bf16_t*       Hout  = (i & 1) ? H1 : H0;
        const int     K     = (i == 0) ? 128 : EMB;
        const bf16_t* Wl    = (i == 0) ? Wbf : (Wbf + W0_ELEMS + (size_t)(i - 1) * EMB * EMB);
        const float*  bl    = (i == 0) ? b0 : (bh + (size_t)(i - 1) * EMB);
        const float*  aScl  = (i == 0) ? nullptr : (sa + (size_t)(i - 1) * EMB);
        const float*  aShf  = (i == 0) ? nullptr : (sc + (size_t)(i - 1) * EMB);

        mlp_gemm_bf16<<<gemmGrid, 256, 0, stream>>>(Ain, Wl, bl, aScl, aShf, Hout, K);
        mlp_bn_partial<<<statGrid, 128, 0, stream>>>(Hout, psum, psq);
        mlp_bn_finalize<<<EMB / 256, 256, 0, stream>>>(psum, psq,
                                                       gamma + (size_t)i * EMB,
                                                       beta  + (size_t)i * EMB,
                                                       sa + (size_t)i * EMB,
                                                       sc + (size_t)i * EMB);
    }

    // layer 4 output lives in H0 (layers 0,2,4 -> H0)
    mlp_out_gemv<<<NROW / 8, 256, 0, stream>>>(H0,
                                               sa + (size_t)4 * EMB,
                                               sc + (size_t)4 * EMB,
                                               Wout, bout, out);
}
